// Encoder_7430293422327
// MI455X (gfx1250) — matrix operand order repaired and checked by reading, not compiled, not run on hardware
//
#include <hip/hip_runtime.h>

// ---------------------------------------------------------------------------
// GCN encoder for MI455X (gfx1250, wave32, WMMA + async global->LDS).
//   h1  = ReLU( GCN(x, W1, b1) )            256 -> 64
//   mu  = GCN(h1, W_mu, b_mu)                64 -> 32   \ fused into one
//   ls  = GCN(h1, W_ls, b_ls)                64 -> 32   / 64-wide pass
// GEMMs: v_wmma_f32_16x16x32_bf16, X tiles staged with
// global_load_async_to_lds_b128 (ASYNCcnt), weights staged bf16-transposed.
// Edge aggregation: fp32 HW atomics on an L2-resident accumulator.
// ---------------------------------------------------------------------------

#define NN 100000
#define NE 1600000
#define INCH 256
#define HID 64

typedef __attribute__((ext_vector_type(16))) __bf16 v16bf;
typedef __attribute__((ext_vector_type(4)))  __bf16 v4bf;
typedef __attribute__((ext_vector_type(8)))  float  v8f;
typedef __attribute__((ext_vector_type(4)))  float  v4f;

__device__ __forceinline__ unsigned short f2bf(float f) {
  __bf16 b = (__bf16)f;                       // RNE, native cvt when available
  return __builtin_bit_cast(unsigned short, b);
}

__device__ __forceinline__ void atomAddF(float* p, float v) {
  unsafeAtomicAdd(p, v);   // lowers to global_atomic_add_f32 on gfx1250
}

// async copy 16 bytes global -> LDS (per lane), tracked by ASYNCcnt
__device__ __forceinline__ void asyncCopy16(const void* gsrc, void* ldsDst) {
  unsigned lds = (unsigned)(unsigned long long)ldsDst;  // low 32b = LDS offset
  asm volatile("global_load_async_to_lds_b128 %0, %1, off"
               :: "v"(lds), "v"(gsrc) : "memory");
}

__device__ __forceinline__ void waitAsync0() {
  asm volatile("s_wait_asynccnt 0x0" ::: "memory");
}

// ------------------------------ prep kernels -------------------------------

__global__ void deg_init_kernel(unsigned int* cnt, int n) {
  int i = blockIdx.x * 256 + threadIdx.x;
  if (i < n) cnt[i] = 0u;
}

__global__ void deg_accum_kernel(const long long* __restrict__ dst,
                                 unsigned int* __restrict__ cnt, int ne) {
  int e = blockIdx.x * 256 + threadIdx.x;
  if (e < ne) atomicAdd(&cnt[(long long)dst[e]], 1u);
}

__global__ void dinv_kernel(const unsigned int* __restrict__ cnt,
                            float* __restrict__ dinv, int n) {
  int i = blockIdx.x * 256 + threadIdx.x;
  if (i < n) dinv[i] = rsqrtf(1.0f + (float)cnt[i]);
}

__global__ void enorm_kernel(const long long* __restrict__ src,
                             const long long* __restrict__ dst,
                             const float* __restrict__ dinv,
                             float* __restrict__ en, int ne) {
  int e = blockIdx.x * 256 + threadIdx.x;
  if (e < ne) en[e] = dinv[(long long)src[e]] * dinv[(long long)dst[e]];
}

__global__ void zero4_kernel(float4* __restrict__ p, int n4) {
  int i = blockIdx.x * 256 + threadIdx.x;
  if (i < n4) p[i] = make_float4(0.f, 0.f, 0.f, 0.f);
}

// --------------------------- WMMA fragment utils ---------------------------
// A fragment (16x32 bf16): lanes 0-15 row M=lane, K kb..kb+7 and kb+16..kb+23
// with kb=0; lanes 16-31 same rows, kb=8.  Assembled from an fp32 LDS tile
// using vector fp32->bf16 converts (v_cvt_pk_bf16_f32 capable).

__device__ __forceinline__ v16bf makeAfrag(const float* ap /*16B aligned*/) {
  union { v16bf v; v4bf b[4]; } ua;
  ua.b[0] = __builtin_convertvector(*(const v4f*)(ap),      v4bf);  // kb+0..3
  ua.b[1] = __builtin_convertvector(*(const v4f*)(ap + 4),  v4bf);  // kb+4..7
  ua.b[2] = __builtin_convertvector(*(const v4f*)(ap + 16), v4bf);  // kb+16..19
  ua.b[3] = __builtin_convertvector(*(const v4f*)(ap + 20), v4bf);  // kb+20..23
  return ua.v;
}

// ------------------------------ WMMA GEMM 1 --------------------------------
// H[N,64] = X[N,256] @ W[256,64].  Block = 256 threads = 8 waves;
// block tile = 32 rows x 64 cols; each wave owns one 16x16 D tile.
// N is an exact multiple of 32 -> no row guards anywhere.

__global__ __launch_bounds__(256) void gemm1_kernel(
    const float* __restrict__ X, const float* __restrict__ W,
    float* __restrict__ H) {
  constexpr int K = INCH;
  constexpr int WSTR = 264;  // bf16 elems per Wt row (528 B, 16B-aligned)
  constexpr int XFS  = 36;   // fp32 elems per Xf row (144 B, 16B-aligned)
  __shared__ __align__(16) unsigned short Wt[64 * WSTR];  // [col][k] bf16
  __shared__ __align__(16) float          Xf[32 * XFS];   // [row][k mod 32]

  const int t = threadIdx.x;
  for (int idx = t; idx < K * 64; idx += 256) {   // transpose W into LDS
    int k = idx >> 6, n = idx & 63;
    Wt[n * WSTR + k] = f2bf(W[idx]);
  }

  const int wave  = t >> 5, lane = t & 31;
  const int rtile = wave >> 2, ctile = wave & 3;
  const int row0  = blockIdx.x * 32;
  const int lrow  = lane & 15;
  const int khalf = lane >> 4;           // 0: lanes 0-15, 1: lanes 16-31
  const int xr    = t >> 3;              // staging row   0..31
  const int xk    = (t & 7) << 2;        // staging k     0,4,..,28

  v8f acc = {};
#pragma unroll
  for (int ks = 0; ks < K; ks += 32) {
    __syncthreads();                      // previous tile fully consumed
    asyncCopy16(X + (long long)(row0 + xr) * K + ks + xk, &Xf[xr * XFS + xk]);
    waitAsync0();
    __syncthreads();                      // tile visible to all waves

    v16bf a = makeAfrag(&Xf[(rtile * 16 + lrow) * XFS + khalf * 8]);

    // B fragment: 32x16 bf16. lanes 0-15: K 0..7,16..23 ; lanes 16-31: K 8..15,24..31.
    union { v16bf v; uint4 q[2]; } ub;
    const unsigned short* bp =
        &Wt[(ctile * 16 + lrow) * WSTR + ks + khalf * 8];
    ub.q[0] = *(const uint4*)(bp);
    ub.q[1] = *(const uint4*)(bp + 16);

    acc = __builtin_amdgcn_wmma_f32_16x16x32_bf16(false, a, false, ub.v,
                                                  (short)0, acc, false, false);
  }

  const int col = ctile * 16 + lrow;
  float* hp = H + (long long)(row0 + rtile * 16 + khalf * 8) * 64 + col;
#pragma unroll
  for (int r = 0; r < 8; ++r) hp[r * 64] = acc[r];
}

// ------------------------------ WMMA GEMM 2 --------------------------------
// H2[N,64] = H1[N,64] @ [W_mu | W_ls]  (weights concatenated along cols).

__global__ __launch_bounds__(256) void gemm2_kernel(
    const float* __restrict__ Hin, const float* __restrict__ Wmu,
    const float* __restrict__ Wls, float* __restrict__ Hout) {
  constexpr int K = HID;
  constexpr int WSTR = 72;   // 144 B rows, 16B-aligned
  constexpr int XFS  = 36;
  __shared__ __align__(16) unsigned short Wt[64 * WSTR];
  __shared__ __align__(16) float          Xf[32 * XFS];

  const int t = threadIdx.x;
  for (int idx = t; idx < K * 64; idx += 256) {
    int k = idx >> 6, n = idx & 63;
    float w = (n < 32) ? Wmu[k * 32 + n] : Wls[k * 32 + (n - 32)];
    Wt[n * WSTR + k] = f2bf(w);
  }

  const int wave  = t >> 5, lane = t & 31;
  const int rtile = wave >> 2, ctile = wave & 3;
  const int row0  = blockIdx.x * 32;
  const int lrow  = lane & 15;
  const int khalf = lane >> 4;
  const int xr    = t >> 3;
  const int xk    = (t & 7) << 2;

  v8f acc = {};
#pragma unroll
  for (int ks = 0; ks < K; ks += 32) {
    __syncthreads();
    asyncCopy16(Hin + (long long)(row0 + xr) * K + ks + xk,
                &Xf[xr * XFS + xk]);
    waitAsync0();
    __syncthreads();

    v16bf a = makeAfrag(&Xf[(rtile * 16 + lrow) * XFS + khalf * 8]);

    union { v16bf v; uint4 q[2]; } ub;
    const unsigned short* bp =
        &Wt[(ctile * 16 + lrow) * WSTR + ks + khalf * 8];
    ub.q[0] = *(const uint4*)(bp);
    ub.q[1] = *(const uint4*)(bp + 16);

    acc = __builtin_amdgcn_wmma_f32_16x16x32_bf16(false, a, false, ub.v,
                                                  (short)0, acc, false, false);
  }

  const int col = ctile * 16 + lrow;
  float* hp = Hout + (long long)(row0 + rtile * 16 + khalf * 8) * 64 + col;
#pragma unroll
  for (int r = 0; r < 8; ++r) hp[r * 64] = acc[r];
}

// --------------------------- edge aggregation ------------------------------
// agg[dst] += h[src] * enorm[e], F=64. 16 threads/edge, float4 gathers,
// fp32 HW atomics (h and agg are 25.6 MB each -> L2-resident on 192 MB L2).

__global__ void edge_agg_kernel(const float* __restrict__ h,
                                const long long* __restrict__ src,
                                const long long* __restrict__ dst,
                                const float* __restrict__ en,
                                float* __restrict__ agg, int ne) {
  int t = blockIdx.x * 256 + threadIdx.x;
  int e = t >> 4;
  if (e >= ne) return;
  int c4 = (t & 15) << 2;
  long long s = src[e], d = dst[e];
  float w = en[e];
  const float4 v = *(const float4*)(h + s * 64 + c4);
  float* a = agg + d * 64 + c4;
  atomAddF(a + 0, v.x * w);
  atomAddF(a + 1, v.y * w);
  atomAddF(a + 2, v.z * w);
  atomAddF(a + 3, v.w * w);
}

// ------------------------------- combines ----------------------------------

__global__ void combine1_kernel(const float* agg,   // may alias h1 (in-place)
                                const float* __restrict__ h,
                                const float* __restrict__ dinv,
                                const float* __restrict__ b1,
                                float* h1, int n) {
  int idx = blockIdx.x * 256 + threadIdx.x;
  if (idx >= n) return;
  int i = idx >> 6, c = idx & 63;
  float di = dinv[i];
  float v = agg[idx] + h[idx] * (di * di) + b1[c];
  h1[idx] = fmaxf(v, 0.0f);   // same-index in-place over agg is safe
}

__global__ void combine2_kernel(const float* __restrict__ agg,
                                const float* __restrict__ h2,
                                const float* __restrict__ dinv,
                                const float* __restrict__ bmu,
                                const float* __restrict__ bls,
                                float* __restrict__ out, int n) {
  int idx = blockIdx.x * 256 + threadIdx.x;
  if (idx >= n) return;
  int i = idx >> 6, c = idx & 63;
  float di = dinv[i];
  float v = agg[idx] + h2[idx] * (di * di);
  if (c < 32)
    out[(long long)i * 32 + c] = v + bmu[c];                       // mu
  else
    out[(long long)NN * 32 + (long long)i * 32 + (c - 32)] = v + bls[c - 32];
}

// ------------------------------- launcher ----------------------------------

extern "C" void kernel_launch(void* const* d_in, const int* in_sizes, int n_in,
                              void* d_out, int out_size, void* d_ws,
                              size_t ws_size, hipStream_t stream) {
  (void)in_sizes; (void)n_in; (void)out_size; (void)ws_size;
  const float*     x    = (const float*)d_in[0];
  const long long* ei   = (const long long*)d_in[1];
  const float*     W1   = (const float*)d_in[2];
  const float*     b1   = (const float*)d_in[3];
  const float*     Wmu  = (const float*)d_in[4];
  const float*     bmu  = (const float*)d_in[5];
  const float*     Wls  = (const float*)d_in[6];
  const float*     bls  = (const float*)d_in[7];
  float*           out  = (float*)d_out;

  const long long* src = ei;
  const long long* dst = ei + NE;

  // workspace layout (floats)
  float*        wsf   = (float*)d_ws;
  unsigned int* cnt   = (unsigned int*)wsf;          // [N]
  float*        dinv  = wsf + NN;                    // [N]
  float*        enorm = wsf + 2 * NN;                // [E]
  float*        bufA  = wsf + 2 * NN + NE;           // [N*64]  h, then h2
  float*        bufB  = bufA + (long long)NN * 64;   // [N*64]  agg/h1/agg2

  const int TB = 256;
  const int gN   = (NN + TB - 1) / TB;
  const int gE   = (NE + TB - 1) / TB;
  const int gF   = (NN * 64 + TB - 1) / TB;
  const int gZ4  = (NN * 16 + TB - 1) / TB;   // float4 elements
  const int gAgg = (NE * 16 + TB - 1) / TB;
  const int gMM  = NN / 32;   // 100000 % 32 == 0

  // normalization
  deg_init_kernel<<<gN, TB, 0, stream>>>(cnt, NN);
  deg_accum_kernel<<<gE, TB, 0, stream>>>(dst, cnt, NE);
  dinv_kernel<<<gN, TB, 0, stream>>>(cnt, dinv, NN);
  enorm_kernel<<<gE, TB, 0, stream>>>(src, dst, dinv, enorm, NE);

  // layer 1: 256 -> 64
  gemm1_kernel<<<gMM, TB, 0, stream>>>(x, W1, bufA);
  zero4_kernel<<<gZ4, TB, 0, stream>>>((float4*)bufB, NN * 16);
  edge_agg_kernel<<<gAgg, TB, 0, stream>>>(bufA, src, dst, enorm, bufB, NE);
  combine1_kernel<<<gF, TB, 0, stream>>>(bufB, bufA, dinv, b1, bufB, NN * 64);

  // layer 2 (mu & logstd fused): 64 -> 64
  gemm2_kernel<<<gMM, TB, 0, stream>>>(bufB, Wmu, Wls, bufA);
  zero4_kernel<<<gZ4, TB, 0, stream>>>((float4*)bufB, NN * 16);
  edge_agg_kernel<<<gAgg, TB, 0, stream>>>(bufA, src, dst, enorm, bufB, NE);
  combine2_kernel<<<gF, TB, 0, stream>>>(bufB, bufA, dinv, bmu, bls, out,
                                         NN * 64);
}
